// ManGRU_38826504356367
// MI455X (gfx1250) — matrix operand order repaired and checked by reading, not compiled, not run on hardware
//
#include <hip/hip_runtime.h>
#include <hip/hip_bf16.h>

// ---------------------------------------------------------------------------
// MI455X (gfx1250) 2-layer GRU, B=256 T=2048 I=H=128.
// Latency-bound sequential scan: 16 WGs x 16 batch rows, all weights on-chip
// (9 matrices bf16-fragment-packed in ~310KB dynamic LDS, 3 layer-0 input
// matrices as per-wave register slabs), all GEMMs via v_wmma_f32_16x16x32_bf16.
// ---------------------------------------------------------------------------

typedef __attribute__((ext_vector_type(16))) __bf16 bf16x16;
typedef __attribute__((ext_vector_type(8)))  __bf16 bf16x8;
typedef __attribute__((ext_vector_type(8)))  float  f32x8;

#define TSTEPS       2048
#define HD           128
#define TILE_STRIDE  136            // bf16 elems per tile row (128 + 8 pad => 272B, bank-conflict-free)

// LDS layout (bytes)
#define OFF_W        0
#define W_BYTES      (9 * 32768)    // 9 fragment-packed bf16 128x128 matrices
#define OFF_X        (OFF_W + W_BYTES)
#define XT_BYTES     (16 * TILE_STRIDE * 2)   // 4352B per x tile
#define OFF_H0       (OFF_X + 2 * XT_BYTES)
#define OFF_H1       (OFF_H0 + XT_BYTES)
#define OFF_RH       (OFF_H1 + XT_BYTES)
#define OFF_OUT      (OFF_RH + XT_BYTES)
#define SMEM_BYTES   (OFF_OUT + 64)           // ~316.7 KB < 320 KB WGP LDS

// LDS-resident matrix slots (packed order m=3..11 minus 3)
enum { UZ0 = 0, UR0, UH0, WZ1, WR1, WH1, UZ1, UR1, UH1 };

// ---------------------------------------------------------------------------
// Helpers
// ---------------------------------------------------------------------------
__device__ __forceinline__ f32x8 wmma_bf16(bf16x16 a, bf16x16 b, f32x8 c) {
  // D = A(16x32 bf16) * B(32x16 bf16) + C(16x16 f32)
  return __builtin_amdgcn_wmma_f32_16x16x32_bf16(false, a, false, b, (short)0, c,
                                                 false, false);
}

__device__ __forceinline__ f32x8 splat8(float v) {
  f32x8 r;
#pragma unroll
  for (int i = 0; i < 8; ++i) r[i] = v;
  return r;
}

__device__ __forceinline__ f32x8 sigmoid8(f32x8 a) {
  f32x8 r;
#pragma unroll
  for (int i = 0; i < 8; ++i) r[i] = 1.0f / (1.0f + __expf(-a[i]));  // inf-safe
  return r;
}

__device__ __forceinline__ f32x8 tanh8(f32x8 a) {
  f32x8 r;
#pragma unroll
  for (int i = 0; i < 8; ++i) r[i] = 1.0f - 2.0f / (1.0f + __expf(2.0f * a[i]));
  return r;
}

// A-fragment load: lane L -> row = L&15, hi = L>>4.
// v16bf elems 0..7  = K (kt*32 + hi*8 .. +7)        (VGPR0-3)
//        elems 8..15 = K (kt*32 + 16 + hi*8 .. +7)   (VGPR4-7)
__device__ __forceinline__ void load_a4(const __bf16* tile, int arow, int hi,
                                        bf16x16 A[4]) {
  const __bf16* rp = tile + arow * TILE_STRIDE;
#pragma unroll
  for (int kt = 0; kt < 4; ++kt) {
    bf16x8 lo = *(const bf16x8*)(rp + kt * 32 + hi * 8);
    bf16x8 hh = *(const bf16x8*)(rp + kt * 32 + 16 + hi * 8);
    A[kt] = __builtin_shufflevector(lo, hh, 0, 1, 2, 3, 4, 5, 6, 7, 8, 9, 10,
                                    11, 12, 13, 14, 15);
  }
}

// C-layout (lane holds rows m0..m0+7, one column) -> bf16 LDS tile
__device__ __forceinline__ void store_c(__bf16* tile, int m0, int ncol, f32x8 v) {
#pragma unroll
  for (int j = 0; j < 8; ++j)
    tile[(m0 + j) * TILE_STRIDE + ncol] = (__bf16)v[j];
}

// ---------------------------------------------------------------------------
// Pack kernel: convert 12 fp32 128x128 matrices into bf16 WMMA B-fragment
// layout: dst[((nt*4+kt)*32 + lane)*16 + e] = src[(kt*32 + (e>>3)*16 + (lane>>4)*8 + (e&7))*128
//                                                 + nt*16 + (lane&15)]
// Packed order: 0:Wz0 1:Wr0 2:Wh0 3:Uz0 4:Ur0 5:Uh0 6:Wz1 7:Wr1 8:Wh1 9:Uz1 10:Ur1 11:Uh1
// ---------------------------------------------------------------------------
__global__ void __launch_bounds__(256)
pack_weights(const float* __restrict__ Wz, const float* __restrict__ Wr,
             const float* __restrict__ Wh, const float* __restrict__ Uz,
             const float* __restrict__ Ur, const float* __restrict__ Uh,
             __bf16* __restrict__ wpk) {
  const int m = blockIdx.x;
  const float* srcs[12] = {Wz,         Wr,         Wh,
                           Uz,         Ur,         Uh,
                           Wz + 16384, Wr + 16384, Wh + 16384,
                           Uz + 16384, Ur + 16384, Uh + 16384};
  const float* s = srcs[m];
  __bf16* d = wpk + m * 16384;
  for (int idx = threadIdx.x; idx < 16384; idx += 256) {
    const int e  = idx & 15;
    const int ln = (idx >> 4) & 31;
    const int kt = (idx >> 9) & 3;
    const int nt = (idx >> 11) & 7;
    const int K  = kt * 32 + (e >> 3) * 16 + (ln >> 4) * 8 + (e & 7);
    const int n  = nt * 16 + (ln & 15);
    d[idx] = (__bf16)s[K * HD + n];
  }
}

// ---------------------------------------------------------------------------
// Persistent recurrent kernel: 16 blocks x 256 threads (8 waves).
// Wave w owns output columns [16w,16w+16). 48 WMMA / wave / timestep.
// ---------------------------------------------------------------------------
__global__ void __launch_bounds__(256, 1)
gru_recurrent(const float* __restrict__ x, const float* __restrict__ bz,
              const float* __restrict__ br, const float* __restrict__ bh,
              const float* __restrict__ fcw, const float* __restrict__ fcb,
              const __bf16* __restrict__ wpk, float* __restrict__ out) {
  extern __shared__ float4 smem4[];
  char* smem = (char*)smem4;
  __bf16* h0t = (__bf16*)(smem + OFF_H0);
  __bf16* h1t = (__bf16*)(smem + OFF_H1);
  __bf16* rht = (__bf16*)(smem + OFF_RH);
  float*  oac = (float*)(smem + OFF_OUT);

  const int tid   = threadIdx.x;
  const int wave  = tid >> 5;
  const int lane  = tid & 31;
  const int nloc  = lane & 15;
  const int hi    = lane >> 4;
  const int m0    = hi * 8;
  const int ncol  = wave * 16 + nloc;
  const int btile = blockIdx.x;

  // ---- one-time setup -----------------------------------------------------
  {  // 9 matrices (m=3..11) -> LDS, contiguous float4 copy
    const float4* s = (const float4*)(wpk + 3 * 16384);
    float4* d = (float4*)(smem + OFF_W);
    for (int i = tid; i < W_BYTES / 16; i += 256) d[i] = s[i];
  }
  for (int i = tid; i < 16 * TILE_STRIDE; i += 256) {
    h0t[i] = (__bf16)0.f;
    h1t[i] = (__bf16)0.f;
  }
  if (tid < 16) oac[tid] = 0.f;

  // layer-0 input weight slabs (Wz0,Wr0,Wh0) -> registers (96 VGPRs/lane)
  bf16x16 w0[3][4];
  {
    const char* gb = (const char*)wpk + wave * 4096 + lane * 32;
#pragma unroll
    for (int m = 0; m < 3; ++m)
#pragma unroll
      for (int kt = 0; kt < 4; ++kt)
        w0[m][kt] = *(const bf16x16*)(gb + m * 32768 + kt * 1024);
  }
  const char* wl = smem + OFF_W + wave * 4096 + lane * 32;
#define WFRAG(m, kt) (*(const bf16x16*)(wl + (m) * 32768 + (kt) * 1024))

  const float bz0v = bz[ncol], bz1v = bz[HD + ncol];
  const float br0v = br[ncol], br1v = br[HD + ncol];
  const float bh0v = bh[ncol], bh1v = bh[HD + ncol];

  // x staging: thread -> (row = tid>>4, 8 cols at (tid&15)*8)
  const int xrow = tid >> 4;
  const int xcol = (tid & 15) * 8;
  const float* xrp = x + (size_t)(btile * 16 + xrow) * TSTEPS * HD + xcol;

  {  // preload t=0 tile into xbuf[0]
    float4 a = *(const float4*)(xrp);
    float4 b = *(const float4*)(xrp + 4);
    __bf16* xl = (__bf16*)(smem + OFF_X);
    bf16x8 v;
    v[0] = (__bf16)a.x; v[1] = (__bf16)a.y; v[2] = (__bf16)a.z; v[3] = (__bf16)a.w;
    v[4] = (__bf16)b.x; v[5] = (__bf16)b.y; v[6] = (__bf16)b.z; v[7] = (__bf16)b.w;
    *(bf16x8*)(xl + xrow * TILE_STRIDE + xcol) = v;
  }

  f32x8 h0r = {};  // f32 hidden state, C-layout slab (layer 0)
  f32x8 h1r = {};  // layer 1
  __syncthreads();

  // ---- 2048 sequential timesteps -----------------------------------------
  for (int t = 0; t < TSTEPS; ++t) {
    const int cur = t & 1;
    __bf16* xl = (__bf16*)(smem + OFF_X) + cur * (16 * TILE_STRIDE);
    __bf16* xn = (__bf16*)(smem + OFF_X) + (cur ^ 1) * (16 * TILE_STRIDE);

    // prefetch x(t+1) into registers, hint x(t+2) into cache
    const int tn = (t + 1 < TSTEPS) ? (t + 1) : t;
    float4 pa = *(const float4*)(xrp + (size_t)tn * HD);
    float4 pb = *(const float4*)(xrp + (size_t)tn * HD + 4);
    {
      const int tp = (t + 2 < TSTEPS) ? (t + 2) : (TSTEPS - 1);
      __builtin_prefetch(xrp + (size_t)tp * HD, 0, 1);
    }

    // -------- layer 0, phase 1: z,r pre-acts + x-part of h~ ---------------
    bf16x16 Ax[4], Ah[4];
    load_a4(xl, nloc, hi, Ax);
    load_a4(h0t, nloc, hi, Ah);
    f32x8 az = splat8(bz0v), ar = splat8(br0v), ah = splat8(bh0v);
#pragma unroll
    for (int kt = 0; kt < 4; ++kt) {
      az = wmma_bf16(Ax[kt], w0[0][kt], az);
      ar = wmma_bf16(Ax[kt], w0[1][kt], ar);
      ah = wmma_bf16(Ax[kt], w0[2][kt], ah);
      az = wmma_bf16(Ah[kt], WFRAG(UZ0, kt), az);
      ar = wmma_bf16(Ah[kt], WFRAG(UR0, kt), ar);
    }
    f32x8 z = sigmoid8(az), r = sigmoid8(ar);
    store_c(rht, m0, ncol, r * h0r);
    __syncthreads();

    // -------- layer 0, phase 2: h~ and state update -----------------------
    bf16x16 Ar[4];
    load_a4(rht, nloc, hi, Ar);
#pragma unroll
    for (int kt = 0; kt < 4; ++kt) ah = wmma_bf16(Ar[kt], WFRAG(UH0, kt), ah);
    f32x8 htl = tanh8(ah);
    h0r = (splat8(1.f) - z) * h0r + z * htl;
    store_c(h0t, m0, ncol, h0r);  // h0[t]: next-step state AND layer-1 input
    __syncthreads();

    // -------- layer 1, phase 3: z,r pre-acts + input-part of h~ -----------
    bf16x16 A0[4], A1[4];
    load_a4(h0t, nloc, hi, A0);
    load_a4(h1t, nloc, hi, A1);
    f32x8 az1 = splat8(bz1v), ar1 = splat8(br1v), ah1 = splat8(bh1v);
#pragma unroll
    for (int kt = 0; kt < 4; ++kt) {
      az1 = wmma_bf16(A0[kt], WFRAG(WZ1, kt), az1);
      ar1 = wmma_bf16(A0[kt], WFRAG(WR1, kt), ar1);
      ah1 = wmma_bf16(A0[kt], WFRAG(WH1, kt), ah1);
      az1 = wmma_bf16(A1[kt], WFRAG(UZ1, kt), az1);
      ar1 = wmma_bf16(A1[kt], WFRAG(UR1, kt), ar1);
    }
    f32x8 z1 = sigmoid8(az1), r1 = sigmoid8(ar1);
    store_c(rht, m0, ncol, r1 * h1r);
    __syncthreads();

    // -------- layer 1, phase 4: h~ + update; stage next x tile ------------
    bf16x16 Ar1[4];
    load_a4(rht, nloc, hi, Ar1);
#pragma unroll
    for (int kt = 0; kt < 4; ++kt) ah1 = wmma_bf16(Ar1[kt], WFRAG(UH1, kt), ah1);
    f32x8 ht1 = tanh8(ah1);
    h1r = (splat8(1.f) - z1) * h1r + z1 * ht1;
    store_c(h1t, m0, ncol, h1r);
    {
      bf16x8 v;
      v[0] = (__bf16)pa.x; v[1] = (__bf16)pa.y; v[2] = (__bf16)pa.z; v[3] = (__bf16)pa.w;
      v[4] = (__bf16)pb.x; v[5] = (__bf16)pb.y; v[6] = (__bf16)pb.z; v[7] = (__bf16)pb.w;
      *(bf16x8*)(xn + xrow * TILE_STRIDE + xcol) = v;
    }
    __syncthreads();
  }

  // ---- FC head: out[b] = h1[b,:] . fc_w + fc_b ---------------------------
  const float fw = fcw[ncol];
#pragma unroll
  for (int j = 0; j < 8; ++j) atomicAdd(&oac[m0 + j], h1r[j] * fw);
  __syncthreads();
  if (tid < 16) out[btile * 16 + tid] = oac[tid] + fcb[0];
#undef WFRAG
}

// ---------------------------------------------------------------------------
// Host entry
// ---------------------------------------------------------------------------
extern "C" void kernel_launch(void* const* d_in, const int* in_sizes, int n_in,
                              void* d_out, int out_size, void* d_ws,
                              size_t ws_size, hipStream_t stream) {
  (void)in_sizes; (void)n_in; (void)out_size; (void)ws_size;
  const float* x   = (const float*)d_in[0];
  const float* Wz  = (const float*)d_in[1];
  const float* bz  = (const float*)d_in[2];
  const float* Uz  = (const float*)d_in[3];
  const float* Wr  = (const float*)d_in[4];
  const float* br  = (const float*)d_in[5];
  const float* Ur  = (const float*)d_in[6];
  const float* Wh  = (const float*)d_in[7];
  const float* bh  = (const float*)d_in[8];
  const float* Uh  = (const float*)d_in[9];
  const float* fcw = (const float*)d_in[10];
  const float* fcb = (const float*)d_in[11];
  __bf16* wpk = (__bf16*)d_ws;         // 12 * 32768 B = 384 KB packed weights
  float* out = (float*)d_out;

  pack_weights<<<12, 256, 0, stream>>>(Wz, Wr, Wh, Uz, Ur, Uh, wpk);

  (void)hipFuncSetAttribute((const void*)gru_recurrent,
                            hipFuncAttributeMaxDynamicSharedMemorySize,
                            SMEM_BYTES);
  gru_recurrent<<<16, 256, SMEM_BYTES, stream>>>(x, bz, br, bh, fcw, fcb, wpk,
                                                 out);
}
